// CustomGNN_58402965291534
// MI455X (gfx1250) — matrix operand order repaired and checked by reading, not compiled, not run on hardware
//
#include <hip/hip_runtime.h>
#include <hip/hip_bf16.h>

#define NN 50000
#define EE 800000
#define BB 256
#define CC 128

typedef _Float16 v8h  __attribute__((ext_vector_type(8)));
typedef _Float16 v16h __attribute__((ext_vector_type(16)));
typedef float    v8f  __attribute__((ext_vector_type(8)));

// ---------------------------------------------------------------------------
// WMMA operand loaders (gfx1250 wave32 layouts, CDNA5 ISA §7.12.2)
// A: 16x32 f16.  lane<16: M=lane, halves[0:8]=K(kb+0..7),  halves[8:16]=K(kb+16..23)
//                lane>=16: M=lane-16, halves=K(kb+8..15), K(kb+24..31)
// B: 32x16 f16.  lane<16: N=lane, halves=K(kb+0..7), K(kb+16..23); lane>=16: N=lane-16, K(kb+8..15), K(kb+24..31)
// ---------------------------------------------------------------------------
__device__ __forceinline__ v16h load_a_tile(const _Float16* __restrict__ A, int row0,
                                            int Kpad, int kbase, int lane) {
    int m    = lane & 15;
    int koff = kbase + ((lane >> 4) << 3);
    const _Float16* p = A + (size_t)(row0 + m) * Kpad + koff;
    v8h lo = *(const v8h*)p;
    v8h hi = *(const v8h*)(p + 16);
    v16h a;
#pragma unroll
    for (int i = 0; i < 8; ++i) { a[i] = lo[i]; a[i + 8] = hi[i]; }
    return a;
}

__device__ __forceinline__ v16h load_b_tile(const _Float16* __restrict__ Bt, int ncol0,
                                            int Kpad, int kbase, int lane) {
    int n    = ncol0 + (lane & 15);
    int koff = kbase + ((lane >> 4) << 3);
    const _Float16* p = Bt + (size_t)n * Kpad + koff;
    v8h lo = *(const v8h*)p;
    v8h hi = *(const v8h*)(p + 16);
    v16h b;
#pragma unroll
    for (int i = 0; i < 8; ++i) { b[i] = lo[i]; b[i + 8] = hi[i]; }
    return b;
}

// ---------------------------------------------------------------------------
// Conv GEMM: y = relu(A @ W + b) then LayerNorm -> out f32 [N,128]
// block = 256 threads (8 waves), each wave owns one 16x16 n-tile, 16 rows/block
// ---------------------------------------------------------------------------
template <int KT>
__global__ void gemm_ln_kernel(const _Float16* __restrict__ A, const _Float16* __restrict__ Bt,
                               const float* __restrict__ bias, const float* __restrict__ gamma,
                               const float* __restrict__ beta, float* __restrict__ out) {
    __shared__ float tile[16][132];
    __shared__ float rsum[16][17];
    __shared__ float rsq[16][17];
    __shared__ float mrow[16], irow[16];

    const int tid  = threadIdx.x;
    const int lane = tid & 31;
    const int wave = tid >> 5;           // n-tile index 0..7
    const int row0 = blockIdx.x * 16;
    const int Kpad = KT * 32;

    v8f acc = {};
#pragma unroll
    for (int kt = 0; kt < KT; ++kt) {
        v16h a = load_a_tile(A, row0, Kpad, kt * 32, lane);
        v16h b = load_b_tile(Bt, wave * 16, Kpad, kt * 32, lane);
        acc = __builtin_amdgcn_wmma_f32_16x16x32_f16(false, a, false, b,
                                                     (short)0, acc, false, false);
    }
    // D layout: VGPR j -> M = j + (lane>=16 ? 8 : 0), N = lane&15
    const int n  = wave * 16 + (lane & 15);
    const int mb = (lane >> 4) << 3;
    const float bn = bias[n];
#pragma unroll
    for (int j = 0; j < 8; ++j) {
        float v = acc[j] + bn;
        tile[mb + j][n] = v > 0.f ? v : 0.f;
    }
    __syncthreads();

    // LayerNorm over 128 channels per row; 16 threads per row
    const int row = tid >> 4;
    const int sub = tid & 15;
    float s = 0.f, q = 0.f;
#pragma unroll
    for (int j = 0; j < 8; ++j) {
        float v = tile[row][sub * 8 + j];
        s += v; q += v * v;
    }
    rsum[row][sub] = s; rsq[row][sub] = q;
    __syncthreads();
    if (sub == 0) {
        float ts = 0.f, tq = 0.f;
        for (int i = 0; i < 16; ++i) { ts += rsum[row][i]; tq += rsq[row][i]; }
        float m   = ts * (1.f / 128.f);
        float var = tq * (1.f / 128.f) - m * m;
        mrow[row] = m;
        irow[row] = rsqrtf(var + 1e-5f);
    }
    __syncthreads();
    const float m = mrow[row], inv = irow[row];
#pragma unroll
    for (int j = 0; j < 8; ++j) {
        int c = sub * 8 + j;
        float v = tile[row][c];
        out[(size_t)(row0 + row) * 128 + c] = (v - m) * inv * gamma[c] + beta[c];
    }
}

// ---------------------------------------------------------------------------
// FC GEMM: out = relu(A @ W + b), f32 out [N, NC]; BN handled separately.
// ---------------------------------------------------------------------------
template <int KT, int NC>
__global__ void gemm_relu_kernel(const _Float16* __restrict__ A, const _Float16* __restrict__ Bt,
                                 const float* __restrict__ bias, float* __restrict__ out) {
    const int tid  = threadIdx.x;
    const int lane = tid & 31;
    const int wave = tid >> 5;
    const int row0 = blockIdx.x * 16;
    const int Kpad = KT * 32;
    const int NT   = NC / 16;

    for (int t = wave; t < NT; t += 8) {
        v8f acc = {};
#pragma unroll
        for (int kt = 0; kt < KT; ++kt) {
            v16h a = load_a_tile(A, row0, Kpad, kt * 32, lane);
            v16h b = load_b_tile(Bt, t * 16, Kpad, kt * 32, lane);
            acc = __builtin_amdgcn_wmma_f32_16x16x32_f16(false, a, false, b,
                                                         (short)0, acc, false, false);
        }
        const int n  = t * 16 + (lane & 15);
        const int mb = (lane >> 4) << 3;
        const float bn = bias[n];
#pragma unroll
        for (int j = 0; j < 8; ++j) {
            float v = acc[j] + bn;
            out[(size_t)(row0 + mb + j) * NC + n] = v > 0.f ? v : 0.f;
        }
    }
}

// ---------------------------------------------------------------------------
// Elementwise / edge / reduction kernels
// ---------------------------------------------------------------------------
__global__ void zero_f32_kernel(float* p, int count) {
    int i = blockIdx.x * 256 + threadIdx.x;
    if (i < count) p[i] = 0.f;
}

__global__ void prep_weights_kernel(const float* __restrict__ W2, const float* __restrict__ W3,
                                    const float* __restrict__ Wf1, const float* __restrict__ Wf2,
                                    _Float16* W2t, _Float16* W3t, _Float16* Wf1t, _Float16* Wf2t) {
    int i = blockIdx.x * 256 + threadIdx.x;
    if (i < 16384) {                       // 128x128 transposes
        int n = i >> 7, k = i & 127;
        W2t[i] = (_Float16)W2[k * 128 + n];
        W3t[i] = (_Float16)W3[k * 128 + n];
    }
    if (i < 32768) {                       // Wf2 [256,128] -> [128][256]
        int n = i >> 8, k = i & 255;
        Wf2t[i] = (_Float16)Wf2[k * 128 + n];
    }
    if (i < 256 * 416) {                   // Wf1 [385,256] -> [256][416] zero-padded
        int n = i / 416, k = i % 416;
        Wf1t[i] = (_Float16)(k < 385 ? Wf1[k * 256 + n] : 0.f);
    }
}

__global__ void conv1_edge_kernel(const float* __restrict__ x, const int* __restrict__ src,
                                  const int* __restrict__ dst, const float* __restrict__ ea,
                                  float* agg1) {
    int e = blockIdx.x * 256 + threadIdx.x;
    if (e >= EE) return;
    float v = x[src[e]] + ea[e];
    v = v > 0.f ? v : 0.f;
    atomicAdd(agg1 + dst[e], v);
}

__global__ void conv1_node_kernel(const float* __restrict__ x, const float* __restrict__ agg1,
                                  const float* __restrict__ W1, const float* __restrict__ b1,
                                  const float* __restrict__ g, const float* __restrict__ hh,
                                  float* __restrict__ x1) {
    int node = blockIdx.x * 8 + (threadIdx.x >> 5);
    if (node >= NN) return;
    int lane = threadIdx.x & 31;
    float h = x[node] + agg1[node];
    float y[4]; float s = 0.f, q = 0.f;
#pragma unroll
    for (int j = 0; j < 4; ++j) {
        int c = lane * 4 + j;
        float v = h * W1[c] + b1[c];
        v = v > 0.f ? v : 0.f;
        y[j] = v; s += v; q += v * v;
    }
    for (int o = 16; o > 0; o >>= 1) { s += __shfl_xor(s, o); q += __shfl_xor(q, o); }
    float m = s * (1.f / 128.f);
    float inv = rsqrtf(q * (1.f / 128.f) - m * m + 1e-5f);
#pragma unroll
    for (int j = 0; j < 4; ++j) {
        int c = lane * 4 + j;
        x1[(size_t)node * 128 + c] = (y[j] - m) * inv * g[c] + hh[c];
    }
}

// fused: msg = relu(x_prev[src] + ea*We + be); atomic scatter-add into agg[dst]
__global__ void gine_edge_kernel(const float* __restrict__ xprev, const int* __restrict__ src,
                                 const int* __restrict__ dst, const float* __restrict__ ea,
                                 const float* __restrict__ We, const float* __restrict__ be,
                                 float* agg) {
    int e = blockIdx.x * 8 + (threadIdx.x >> 5);
    if (e >= EE) return;
    int lane = threadIdx.x & 31;
    int s = src[e], d = dst[e];
    float a = ea[e];
    const float4 xv = *(const float4*)(xprev + (size_t)s * 128 + lane * 4);
    const float4 wv = *(const float4*)(We + lane * 4);
    const float4 bv = *(const float4*)(be + lane * 4);
    float* ap = agg + (size_t)d * 128 + lane * 4;
    float m0 = xv.x + a * wv.x + bv.x; m0 = m0 > 0.f ? m0 : 0.f;
    float m1 = xv.y + a * wv.y + bv.y; m1 = m1 > 0.f ? m1 : 0.f;
    float m2 = xv.z + a * wv.z + bv.z; m2 = m2 > 0.f ? m2 : 0.f;
    float m3 = xv.w + a * wv.w + bv.w; m3 = m3 > 0.f ? m3 : 0.f;
    atomicAdd(ap + 0, m0); atomicAdd(ap + 1, m1);
    atomicAdd(ap + 2, m2); atomicAdd(ap + 3, m3);
}

__global__ void hbuild_kernel(const float* __restrict__ xp, const float* __restrict__ agg,
                              _Float16* __restrict__ hb, int count) {
    int i = blockIdx.x * 256 + threadIdx.x;
    if (i < count) hb[i] = (_Float16)(xp[i] + agg[i]);
}

__global__ void xcpad_kernel(const float* __restrict__ x, const float* __restrict__ x1,
                             const float* __restrict__ x2, const float* __restrict__ x3,
                             _Float16* __restrict__ xc) {
    int i = blockIdx.x * 256 + threadIdx.x;
    if (i >= NN * 416) return;
    int node = i / 416, c = i % 416;
    float v;
    if (c == 0)        v = x[node];
    else if (c < 129)  v = x1[(size_t)node * 128 + (c - 1)];
    else if (c < 257)  v = x2[(size_t)node * 128 + (c - 129)];
    else if (c < 385)  v = x3[(size_t)node * 128 + (c - 257)];
    else               v = 0.f;
    xc[i] = (_Float16)v;
}

template <int NC>
__global__ void bn_stats_kernel(const float* __restrict__ y, float* sum, float* sq, int nrows) {
    int tid = threadIdx.x;
    int c = tid % NC;
    int g = tid / NC;
    const int G = 256 / NC;
    int r0 = blockIdx.x * 32;
    float s = 0.f, q = 0.f;
    for (int r = r0 + g; r < r0 + 32 && r < nrows; r += G) {
        float v = y[(size_t)r * NC + c];
        s += v; q += v * v;
    }
    atomicAdd(&sum[c], s);
    atomicAdd(&sq[c], q);
}

template <int NC>
__global__ void bn_apply_kernel(const float* __restrict__ y, const float* __restrict__ sum,
                                const float* __restrict__ sq, const float* __restrict__ g,
                                const float* __restrict__ b, float* out32, _Float16* out16,
                                int nrows) {
    int i = blockIdx.x * 256 + threadIdx.x;
    if (i >= nrows * NC) return;
    int c = i % NC;
    float m   = sum[c] * (1.f / (float)nrows);
    float var = sq[c] * (1.f / (float)nrows) - m * m;
    float inv = rsqrtf(var + 1e-5f);
    float v = (y[i] - m) * inv * g[c] + b[c];
    if (out32) out32[i] = v;
    if (out16) out16[i] = (_Float16)v;
}

__global__ void edge_head_kernel(const float* __restrict__ l2, const int* __restrict__ src,
                                 const float* __restrict__ Wp, const float* __restrict__ bp,
                                 float* __restrict__ out) {
    int e = blockIdx.x * 8 + (threadIdx.x >> 5);
    if (e >= EE) return;
    int lane = threadIdx.x & 31;
    const float4 v = *(const float4*)(l2 + (size_t)src[e] * 128 + lane * 4);
    const float4 w = *(const float4*)(Wp + lane * 4);
    float s = v.x * w.x + v.y * w.y + v.z * w.z + v.w * w.w;
    for (int o = 16; o > 0; o >>= 1) s += __shfl_xor(s, o);
    if (lane == 0) out[e] = 1.f / (1.f + __expf(-(s + bp[0])));
}

__global__ void pool_scatter_kernel(const float* __restrict__ l2, const int* __restrict__ batch,
                                    float* pooled, float* cnt) {
    int node = blockIdx.x * 8 + (threadIdx.x >> 5);
    if (node >= NN) return;
    int lane = threadIdx.x & 31;
    int b = batch[node];
    const float4 v = *(const float4*)(l2 + (size_t)node * 128 + lane * 4);
    float* q = pooled + (size_t)b * 128 + lane * 4;
    atomicAdd(q + 0, v.x); atomicAdd(q + 1, v.y);
    atomicAdd(q + 2, v.z); atomicAdd(q + 3, v.w);
    if (lane == 0) atomicAdd(cnt + b, 1.f);
}

__global__ void value_head_kernel(const float* __restrict__ pooled, const float* __restrict__ cnt,
                                  const float* __restrict__ Wv, const float* __restrict__ bv,
                                  float* __restrict__ out) {
    int b = blockIdx.x * 8 + (threadIdx.x >> 5);
    if (b >= BB) return;
    int lane = threadIdx.x & 31;
    float n = fmaxf(cnt[b], 1.f);
    const float4 v = *(const float4*)(pooled + (size_t)b * 128 + lane * 4);
    const float4 w = *(const float4*)(Wv + lane * 4);
    float s = (v.x * w.x + v.y * w.y + v.z * w.z + v.w * w.w);
    for (int o = 16; o > 0; o >>= 1) s += __shfl_xor(s, o);
    if (lane == 0) out[b] = tanhf(s / n + bv[0]);
}

// ---------------------------------------------------------------------------
extern "C" void kernel_launch(void* const* d_in, const int* in_sizes, int n_in,
                              void* d_out, int out_size, void* d_ws, size_t ws_size,
                              hipStream_t stream) {
    const float* x      = (const float*)d_in[0];
    const int*   ei     = (const int*)d_in[1];
    const int*   src    = ei;
    const int*   dst    = ei + EE;
    const float* ea     = (const float*)d_in[2];
    const int*   batch  = (const int*)d_in[3];
    const float* W1  = (const float*)d_in[4];  const float* b1  = (const float*)d_in[5];
    const float* g1  = (const float*)d_in[6];  const float* h1  = (const float*)d_in[7];
    const float* We2 = (const float*)d_in[8];  const float* be2 = (const float*)d_in[9];
    const float* W2  = (const float*)d_in[10]; const float* b2  = (const float*)d_in[11];
    const float* g2  = (const float*)d_in[12]; const float* h2  = (const float*)d_in[13];
    const float* We3 = (const float*)d_in[14]; const float* be3 = (const float*)d_in[15];
    const float* W3  = (const float*)d_in[16]; const float* b3  = (const float*)d_in[17];
    const float* g3  = (const float*)d_in[18]; const float* h3  = (const float*)d_in[19];
    const float* Wf1 = (const float*)d_in[20]; const float* bf1 = (const float*)d_in[21];
    const float* gb1 = (const float*)d_in[22]; const float* hb1 = (const float*)d_in[23];
    const float* Wf2 = (const float*)d_in[24]; const float* bf2 = (const float*)d_in[25];
    const float* gb2 = (const float*)d_in[26]; const float* hb2 = (const float*)d_in[27];
    const float* Wp  = (const float*)d_in[28]; const float* bp  = (const float*)d_in[29];
    const float* Wv  = (const float*)d_in[30]; const float* bv  = (const float*)d_in[31];
    float* out = (float*)d_out;

    // ---- workspace carve-up (256B aligned) ----
    char* base = (char*)d_ws;
    size_t off = 0;
    auto alloc = [&](size_t bytes) { size_t o = off; off = (off + bytes + 255) & ~(size_t)255; return o; };
    const size_t NODE32 = (size_t)NN * 128 * 4;
    size_t o_x1   = alloc(NODE32);                 // f32 [N,128]   (later: l1pre part 1)
    size_t o_x2   = alloc(NODE32);                 // f32 [N,128]   (later: l1pre part 2)
    size_t o_x3   = alloc(NODE32);                 // f32 [N,128]   (later: l1h f16 [N,256])
    size_t o_agg  = alloc(NODE32);                 // f32 [N,128]   (later: l2pre)
    size_t o_hbuf = alloc((size_t)NN * 128 * 2);   // f16 [N,128]
    size_t o_xc   = alloc((size_t)NN * 416 * 2);   // f16 [N,416]   (later: l2 f32 [N,128])
    size_t o_agg1 = alloc((size_t)NN * 4);
    size_t o_W2t  = alloc(128 * 128 * 2);
    size_t o_W3t  = alloc(128 * 128 * 2);
    size_t o_Wf2t = alloc(128 * 256 * 2);
    size_t o_Wf1t = alloc(256 * 416 * 2);
    size_t o_st1  = alloc(512 * 4);                // sum1[256], sq1[256]
    size_t o_st2  = alloc(256 * 4);                // sum2[128], sq2[128]
    size_t o_pool = alloc((size_t)BB * 128 * 4);
    size_t o_cnt  = alloc((size_t)BB * 4);
    (void)ws_size; (void)in_sizes; (void)n_in; (void)out_size;

    float*    x1f   = (float*)(base + o_x1);
    float*    x2f   = (float*)(base + o_x2);
    float*    x3f   = (float*)(base + o_x3);
    float*    aggf  = (float*)(base + o_agg);
    _Float16* hbuf  = (_Float16*)(base + o_hbuf);
    _Float16* xcpad = (_Float16*)(base + o_xc);
    float*    agg1  = (float*)(base + o_agg1);
    _Float16* W2t   = (_Float16*)(base + o_W2t);
    _Float16* W3t   = (_Float16*)(base + o_W3t);
    _Float16* Wf2t  = (_Float16*)(base + o_Wf2t);
    _Float16* Wf1t  = (_Float16*)(base + o_Wf1t);
    float*    sum1  = (float*)(base + o_st1);  float* sq1 = sum1 + 256;
    float*    sum2  = (float*)(base + o_st2);  float* sq2 = sum2 + 128;
    float*    pooled= (float*)(base + o_pool);
    float*    cnt   = (float*)(base + o_cnt);
    // reuse aliases
    float*    l1pre = x1f;                       // [N,256] f32 = x1+x2 regions
    _Float16* l1h   = (_Float16*)x3f;            // [N,256] f16 = x3 region
    float*    l2pre = aggf;                      // [N,128] f32
    float*    l2    = (float*)xcpad;             // [N,128] f32

    const int T = 256;
    auto blocks = [](long long n, int t) { return (int)((n + t - 1) / t); };

    // 0) weight prep (transpose + f16, pad fc1 K to 416)
    prep_weights_kernel<<<blocks(256 * 416, T), T, 0, stream>>>(W2, W3, Wf1, Wf2, W2t, W3t, Wf1t, Wf2t);

    // 1) conv1 (F=1 scalar path)
    zero_f32_kernel<<<blocks(NN, T), T, 0, stream>>>(agg1, NN);
    conv1_edge_kernel<<<blocks(EE, T), T, 0, stream>>>(x, src, dst, ea, agg1);
    conv1_node_kernel<<<NN / 8, T, 0, stream>>>(x, agg1, W1, b1, g1, h1, x1f);

    // 2) conv2: fused edge (ea@We2+be2 in-register) -> agg; h = x1+agg (f16); WMMA GEMM + LN
    zero_f32_kernel<<<blocks((long long)NN * 128, T), T, 0, stream>>>(aggf, NN * 128);
    gine_edge_kernel<<<EE / 8, T, 0, stream>>>(x1f, src, dst, ea, We2, be2, aggf);
    hbuild_kernel<<<blocks((long long)NN * 128, T), T, 0, stream>>>(x1f, aggf, hbuf, NN * 128);
    gemm_ln_kernel<4><<<NN / 16, T, 0, stream>>>(hbuf, W2t, b2, g2, h2, x2f);

    // 3) conv3
    zero_f32_kernel<<<blocks((long long)NN * 128, T), T, 0, stream>>>(aggf, NN * 128);
    gine_edge_kernel<<<EE / 8, T, 0, stream>>>(x2f, src, dst, ea, We3, be3, aggf);
    hbuild_kernel<<<blocks((long long)NN * 128, T), T, 0, stream>>>(x2f, aggf, hbuf, NN * 128);
    gemm_ln_kernel<4><<<NN / 16, T, 0, stream>>>(hbuf, W3t, b3, g3, h3, x3f);

    // 4) concat [x|x1|x2|x3] zero-padded to K=416 (f16)
    xcpad_kernel<<<blocks((long long)NN * 416, T), T, 0, stream>>>(x, x1f, x2f, x3f, xcpad);

    // 5) fc1: WMMA GEMM (K=416, NC=256) + relu, then BatchNorm (stats + apply -> f16)
    gemm_relu_kernel<13, 256><<<NN / 16, T, 0, stream>>>(xcpad, Wf1t, bf1, l1pre);
    zero_f32_kernel<<<blocks(512, T), T, 0, stream>>>(sum1, 512);
    bn_stats_kernel<256><<<blocks(NN, 32), T, 0, stream>>>(l1pre, sum1, sq1, NN);
    bn_apply_kernel<256><<<blocks((long long)NN * 256, T), T, 0, stream>>>(l1pre, sum1, sq1, gb1, hb1,
                                                                            nullptr, l1h, NN);

    // 6) fc2: WMMA GEMM (K=256, NC=128) + relu, then BatchNorm -> l2 f32
    gemm_relu_kernel<8, 128><<<NN / 16, T, 0, stream>>>(l1h, Wf2t, bf2, l2pre);
    zero_f32_kernel<<<blocks(256, T), T, 0, stream>>>(sum2, 256);
    bn_stats_kernel<128><<<blocks(NN, 32), T, 0, stream>>>(l2pre, sum2, sq2, NN);
    bn_apply_kernel<128><<<blocks((long long)NN * 128, T), T, 0, stream>>>(l2pre, sum2, sq2, gb2, hb2,
                                                                            l2, nullptr, NN);

    // 7) heads
    edge_head_kernel<<<EE / 8, T, 0, stream>>>(l2, src, Wp, bp, out);
    zero_f32_kernel<<<blocks(BB * 128 + BB, T), T, 0, stream>>>(pooled, BB * 128 + BB);
    pool_scatter_kernel<<<NN / 8, T, 0, stream>>>(l2, batch, pooled, cnt);
    value_head_kernel<<<BB / 8, T, 0, stream>>>(pooled, cnt, Wv, bv, out + EE);
}
